// LinearResidualMemory_54949811585662
// MI455X (gfx1250) — matrix operand order repaired and checked by reading, not compiled, not run on hardware
//
#include <hip/hip_runtime.h>
#include <hip/hip_bf16.h>

// Problem constants
#define FIN    4096
#define NMEM   256
#define SQF    64
#define BATCH  1024
#define NQK    (SQF * NMEM)   // 16384

typedef __attribute__((ext_vector_type(16))) __bf16 v16bf;
typedef __attribute__((ext_vector_type(8)))  __bf16 v8bf;
typedef __attribute__((ext_vector_type(8)))  float  v8f;

// ---------------------------------------------------------------------------
// Tiled bf16 WMMA GEMM:  C[M,N] = epilogue(A[M,K] @ B[K,N])
//   EPI 0: relu(acc + bias[col])
//   EPI 1: acc
//   EPI 2: acc + bias[col] + resid[row,col]
// Block: 256 threads (8 waves). Tile BM=256 x BN=64, K-step BK=32.
// Each wave owns a 32(M) x 64(N) patch: 2 A-frags x 4 B-frags -> 8 WMMAs
// per K-step. Grid: x = M-tile (fastest) so co-resident blocks share one
// 64-column B stripe (1 MB) in L2; weights stream from HBM exactly once.
// ---------------------------------------------------------------------------
#define BM 256
#define BN 64
#define BK 32
#define LDA_S 40   // BK + 8 bf16 pad (80B rows, 16B-aligned chunks)
#define LDB_S 40

template <int EPI>
__global__ __launch_bounds__(256) void gemm_bf16_wmma(
    const float* __restrict__ A, const float* __restrict__ B,
    const float* __restrict__ bias, const float* __restrict__ resid,
    float* __restrict__ C, int M, int N, int K, int lda, int ldb)
{
    __shared__ __align__(16) __bf16 As[BM * LDA_S];   // 20 KB
    __shared__ __align__(16) __bf16 Bs[BN * LDB_S];   //  5 KB

    const int tid  = threadIdx.x;
    const int lane = tid & 31;
    const int wave = tid >> 5;                  // 0..7 -> 32-row strip
    const int m0   = blockIdx.x * BM;           // M-tile varies fastest
    const int n0   = blockIdx.y * BN;

    const int half = (lane < 16) ? 0 : 1;       // wave32 half select
    const int l15  = lane & 15;

    v8f acc[2][4] = {};                         // [m-subtile][n-subtile]

    for (int k0 = 0; k0 < K; k0 += BK) {
        // ---- stage A: BM x BK f32 -> bf16 LDS (8 float4 per thread) ----
        #pragma unroll
        for (int it = 0; it < 8; ++it) {
            int idx = tid + it * 256;           // 0..2047
            int row = idx >> 3;
            int c4  = (idx & 7) * 4;
            const float4 f = *(const float4*)(A + (size_t)(m0 + row) * lda + k0 + c4);
            __bf16* d = &As[row * LDA_S + c4];
            d[0] = (__bf16)f.x; d[1] = (__bf16)f.y;
            d[2] = (__bf16)f.z; d[3] = (__bf16)f.w;
        }
        // ---- stage B transposed: BK x BN f32 -> Bs[n][k] (2 float4/thread)
        #pragma unroll
        for (int it = 0; it < 2; ++it) {
            int idx = tid + it * 256;           // 0..511
            int kr  = idx >> 4;                 // 0..31
            int c4  = (idx & 15) * 4;           // n offset
            const float4 f = *(const float4*)(B + (size_t)(k0 + kr) * ldb + n0 + c4);
            Bs[(c4 + 0) * LDB_S + kr] = (__bf16)f.x;
            Bs[(c4 + 1) * LDB_S + kr] = (__bf16)f.y;
            Bs[(c4 + 2) * LDB_S + kr] = (__bf16)f.z;
            Bs[(c4 + 3) * LDB_S + kr] = (__bf16)f.w;
        }
        __syncthreads();

        union F { v16bf v; v8bf h[2]; };

        // ---- hoist ALL fragment loads, then run the 8-WMMA chain ----
        // A frag per ISA 16-bit 16x32 layout: lane<16 K{0..7,16..23},
        // lane>=16 K{8..15,24..31}
        F af[2];
        #pragma unroll
        for (int ms = 0; ms < 2; ++ms) {
            const int arow = wave * 32 + ms * 16 + l15;
            const int kA   = half * 8;
            af[ms].h[0] = *(const v8bf*)&As[arow * LDA_S + kA];
            af[ms].h[1] = *(const v8bf*)&As[arow * LDA_S + kA + 16];
        }
        // B frag: fixed N column per lane; lanes 0-15 K{0..7,16..23}, 16-31 K{8..15,24..31}
        F bfr[4];
        #pragma unroll
        for (int ns = 0; ns < 4; ++ns) {
            const int bn = ns * 16 + l15;
            const int kB = half * 8;
            bfr[ns].h[0] = *(const v8bf*)&Bs[bn * LDB_S + kB];
            bfr[ns].h[1] = *(const v8bf*)&Bs[bn * LDB_S + kB + 16];
        }
        #pragma unroll
        for (int ms = 0; ms < 2; ++ms)
            #pragma unroll
            for (int ns = 0; ns < 4; ++ns)
                acc[ms][ns] = __builtin_amdgcn_wmma_f32_16x16x32_bf16(
                    false, af[ms].v, false, bfr[ns].v, (short)0,
                    acc[ms][ns], false, false);
        __syncthreads();
    }

    // ---- epilogue (C/D layout: VGPR r -> M=r (lanes 0-15) / r+8 (16-31)) ----
    #pragma unroll
    for (int ms = 0; ms < 2; ++ms) {
        const int rbase = m0 + wave * 32 + ms * 16 + half * 8;
        #pragma unroll
        for (int ns = 0; ns < 4; ++ns) {
            const int col = n0 + ns * 16 + l15;
            float bv = 0.0f;
            if constexpr (EPI != 1) bv = bias[col];
            #pragma unroll
            for (int r = 0; r < 8; ++r) {
                const int row = rbase + r;
                float v = acc[ms][ns][r];
                if constexpr (EPI == 0) { v += bv; v = v > 0.0f ? v : 0.0f; }
                if constexpr (EPI == 2) { v += bv + resid[(size_t)row * N + col]; }
                C[(size_t)row * N + col] = v;
            }
        }
    }
}

// qsum[b,s] = (1/(256*8)) * sum_m XQ[b, m*64+s]      (XQ already relu'd)
__global__ __launch_bounds__(64) void qsum_kernel(const float* __restrict__ XQ,
                                                  float* __restrict__ qsum)
{
    const int b = blockIdx.x;
    const int s = threadIdx.x;                  // 0..63, coalesced across threads
    const float* p = XQ + (size_t)b * NQK + s;
    float acc = 0.0f;
    #pragma unroll 8
    for (int m = 0; m < NMEM; ++m) acc += p[m * SQF];
    qsum[b * SQF + s] = acc * (1.0f / (256.0f * 8.0f));
}

// sbar[b,n] = sum_s qsum[b,s] * XK[b, n*64+s]        (XK already relu'd)
__global__ __launch_bounds__(256) void sbar_kernel(const float* __restrict__ XK,
                                                   const float* __restrict__ qsum,
                                                   float* __restrict__ sbar)
{
    __shared__ float q[SQF];
    const int b = blockIdx.x;
    const int n = threadIdx.x;                  // 0..255
    if (n < SQF) q[n] = qsum[b * SQF + n];
    __syncthreads();
    const float4* p4 = (const float4*)(XK + (size_t)b * NQK + n * SQF);
    float acc = 0.0f;
    #pragma unroll
    for (int s4 = 0; s4 < SQF / 4; ++s4) {
        const float4 f = p4[s4];
        acc += q[s4 * 4 + 0] * f.x + q[s4 * 4 + 1] * f.y
             + q[s4 * 4 + 2] * f.z + q[s4 * 4 + 3] * f.w;
    }
    sbar[b * NMEM + n] = acc;
}

extern "C" void kernel_launch(void* const* d_in, const int* in_sizes, int n_in,
                              void* d_out, int out_size, void* d_ws, size_t ws_size,
                              hipStream_t stream) {
    const float* x  = (const float*)d_in[0];   // [1024, 4096]
    const float* Wq = (const float*)d_in[1];   // [4096, 16384]
    const float* bq = (const float*)d_in[2];   // [16384]
    const float* Wk = (const float*)d_in[3];   // [4096, 16384]
    const float* bk = (const float*)d_in[4];   // [16384]
    const float* v  = (const float*)d_in[5];   // [256, 4096]
    const float* Wl = (const float*)d_in[6];   // [4096, 4096]
    const float* bl = (const float*)d_in[7];   // [4096]
    float* out = (float*)d_out;                // [1024, 4096]

    // Workspace layout (f32): XQ 64MB | XK 64MB | qsum 256KB | sbar 1MB | pooled 16MB
    float* XQ     = (float*)d_ws;
    float* XK     = XQ + (size_t)BATCH * NQK;
    float* qsum   = XK + (size_t)BATCH * NQK;
    float* sbar   = qsum + (size_t)BATCH * SQF;
    float* pooled = sbar + (size_t)BATCH * NMEM;

    const dim3 blk(256);

    // Grid: x = M-tiles (fastest) for L2 stripe sharing of B.
    // 1) XQ = relu(x @ Wq + bq)   M=1024 N=16384 K=4096
    gemm_bf16_wmma<0><<<dim3(BATCH / BM, NQK / BN), blk, 0, stream>>>(
        x, Wq, bq, nullptr, XQ, BATCH, NQK, FIN, FIN, NQK);
    // 2) XK = relu(x @ Wk + bk)
    gemm_bf16_wmma<0><<<dim3(BATCH / BM, NQK / BN), blk, 0, stream>>>(
        x, Wk, bk, nullptr, XK, BATCH, NQK, FIN, FIN, NQK);
    // 3) qsum over memory slots
    qsum_kernel<<<dim3(BATCH), dim3(64), 0, stream>>>(XQ, qsum);
    // 4) sbar[b,n] = qsum[b,:] . XK[b,n,:]
    sbar_kernel<<<dim3(BATCH), dim3(256), 0, stream>>>(XK, qsum, sbar);
    // 5) pooled = sbar @ v        M=1024 N=4096 K=256
    gemm_bf16_wmma<1><<<dim3(BATCH / BM, FIN / BN), blk, 0, stream>>>(
        sbar, v, nullptr, nullptr, pooled, BATCH, FIN, NMEM, NMEM, FIN);
    // 6) out = x + pooled @ Wl + bl
    gemm_bf16_wmma<2><<<dim3(BATCH / BM, FIN / BN), blk, 0, stream>>>(
        pooled, Wl, bl, x, out, BATCH, FIN, FIN, FIN, FIN);
}
